// ChessformerAttention_62637803044981
// MI455X (gfx1250) — matrix operand order repaired and checked by reading, not compiled, not run on hardware
//
#include <hip/hip_runtime.h>
#include <hip/hip_bf16.h>

// ---------------------------------------------------------------------------
// Chessformer attention for MI455X (gfx1250, wave32, WMMA + TDM).
// Heavy math through v_wmma_f32_16x16x32_bf16 (f32 accumulate); weight tiles
// staged into LDS by the Tensor Data Mover (tensor_load_to_lds, TENSORcnt),
// double-buffered.
// ---------------------------------------------------------------------------

typedef __attribute__((ext_vector_type(16))) __bf16 v16bf;
typedef __attribute__((ext_vector_type(8)))  __bf16 v8bf;
typedef __attribute__((ext_vector_type(4)))  __bf16 v4bf;
typedef __attribute__((ext_vector_type(8)))  float  v8f;

#define D_MODEL 1024
#define NHEADS  16
#define LSEQ    64
#define DHEAD   64
#define BATCH   512
#define MROWS   (BATCH * LSEQ)   // 32768

// ---- TDM availability probe ------------------------------------------------
#if defined(__has_builtin)
#if __has_builtin(__builtin_amdgcn_tensor_load_to_lds) && \
    __has_builtin(__builtin_amdgcn_s_wait_tensorcnt)
#define USE_TDM 1
#endif
#endif
#ifndef USE_TDM
#define USE_TDM 0
#endif

// ---- WMMA fragment loaders (contiguous k = 16*hi + s, same order for A and B) ------
// 16-bit A matrix 16x32: lanes 0-15 hold K=0..15, lanes 16-31 hold K=16..31.
static __device__ __forceinline__ v16bf frag_a_bf16(const __bf16* base, int row,
                                                    int ld, int kb, int lane) {
  const int hi = lane >> 4;
  const __bf16* p = base + (size_t)row * ld + kb + (hi << 4);
  v8bf lo = *(const v8bf*)(p);
  v8bf hh = *(const v8bf*)(p + 8);
  return __builtin_shufflevector(lo, hh, 0, 1, 2, 3, 4, 5, 6, 7,
                                 8, 9, 10, 11, 12, 13, 14, 15);
}

// 16-bit B matrix 32x16: lanes 0-15 hold K=0..15, lanes 16-31 hold K=16..31.
// "col" indexes the N dimension; source is row-major [N, K] (pre-transposed).
static __device__ __forceinline__ v16bf frag_b_bf16(const __bf16* base, int col,
                                                    int ld, int kb, int lane) {
  const int hi = lane >> 4;
  const __bf16* p = base + (size_t)col * ld + kb + (hi << 4);
  v8bf lo = *(const v8bf*)(p);
  v8bf hh = *(const v8bf*)(p + 8);
  return __builtin_shufflevector(lo, hh, 0, 1, 2, 3, 4, 5, 6, 7,
                                 8, 9, 10, 11, 12, 13, 14, 15);
}

static __device__ __forceinline__ v8f wmma_bf16(v16bf a, v16bf b, v8f c) {
  return __builtin_amdgcn_wmma_f32_16x16x32_bf16(false, a, false, b,
                                                 (short)0, c, false, false);
}

// ---- TDM: DMA a 64(n) x 32(k) bf16 tile of Bt[N=1024,K=1024] into LDS -----
#if USE_TDM
typedef __attribute__((ext_vector_type(4))) unsigned int tdm_v4u;
typedef __attribute__((ext_vector_type(8))) int          tdm_v8i;
typedef __attribute__((ext_vector_type(4))) int          tdm_v4i;

static __device__ __forceinline__ void tdm_load_b_tile(const __bf16* gsrc,
                                                       unsigned int lds_off) {
  const unsigned long long ga = (unsigned long long)(uintptr_t)gsrc;
  // D# group 0 (CDNA5 ISA): count=1, lds_addr,
  // global_addr[56:0], type=2 ("image").
  tdm_v4u g0;
  g0[0] = 1u;
  g0[1] = lds_off;
  g0[2] = (unsigned int)ga;
  g0[3] = (unsigned int)(ga >> 32) | (2u << 30);
  // D# group 1 (§8.4): data_size=2B; tensor_dim0=1024, tensor_dim1=1024;
  // tile_dim0=32 (k), tile_dim1=64 (n); tensor_dim0_stride=1024.
  tdm_v8i g1;
  g1[0] = (int)(1u << 16);            // workgroup_mask=0, data_size=1 (2B)
  g1[1] = (int)(1024u << 16);         // tensor_dim0[15:0]=1024
  g1[2] = (int)(1024u << 16);         // tensor_dim0[31:16]=0, tensor_dim1[15:0]=1024
  g1[3] = (int)(32u << 16);           // tensor_dim1[31:16]=0, tile_dim0=32
  g1[4] = 64;                         // tile_dim1=64, tile_dim2=0
  g1[5] = 1024;                       // tensor_dim0_stride[31:0]
  g1[6] = 0;                          // stride0[47:32]=0, dim1_stride[15:0]=0
  g1[7] = 0;
  tdm_v4i z4 = (tdm_v4i)0;
#if __clang_major__ >= 23
  tdm_v8i z8 = (tdm_v8i)0;
  __builtin_amdgcn_tensor_load_to_lds(g0, g1, z4, z4, z8, 0);
#else
  __builtin_amdgcn_tensor_load_to_lds(g0, g1, z4, z4, 0);
#endif
}
#endif  // USE_TDM

// ---- x f32 -> bf16, one-time pass (keeps GEMM A loads as b128) ------------
__global__ __launch_bounds__(256) void convert_x_kernel(
    const float* __restrict__ src, __bf16* __restrict__ dst) {
  const size_t i = ((size_t)blockIdx.x * 256 + threadIdx.x) * 4;
  float4 v = *(const float4*)(src + i);
  v4bf o;
  o[0] = (__bf16)v.x; o[1] = (__bf16)v.y;
  o[2] = (__bf16)v.z; o[3] = (__bf16)v.w;
  *(v4bf*)(dst + i) = o;
}

// ---- W [K,N] f32 -> Wt [N,K] bf16, LDS-tiled transpose --------------------
__global__ __launch_bounds__(256) void transpose_convert_kernel(
    const float* __restrict__ src, __bf16* __restrict__ dst) {
  __shared__ float tile[32][33];
  int x = blockIdx.x * 32 + threadIdx.x;  // src col (n)
  int y = blockIdx.y * 32 + threadIdx.y;  // src row (k)
#pragma unroll
  for (int j = 0; j < 32; j += 8)
    tile[threadIdx.y + j][threadIdx.x] = src[(size_t)(y + j) * D_MODEL + x];
  __syncthreads();
  int xo = blockIdx.y * 32 + threadIdx.x;  // dst col (k)
  int yo = blockIdx.x * 32 + threadIdx.y;  // dst row (n)
#pragma unroll
  for (int j = 0; j < 32; j += 8)
    dst[(size_t)(yo + j) * D_MODEL + xo] =
        (__bf16)tile[threadIdx.x][threadIdx.y + j];
}

// ---- Generic GEMM: C[M,N] = A[M,K] @ Bt[N,K]^T ----------------------------
// MODE 0: store bf16 as [B, H, L, Dh]   (Q, K)
// MODE 1: store bf16 as [B, H, Dh, L]   (V transposed for the PV matmul)
// MODE 2: store f32 row-major [M, N]    (final output)
// 128 threads (4 waves), 64x64 block tile; B tiles TDM-staged in LDS with
// double buffering; A fragments straight from global (b128, no redundancy).
template <int MODE>
__global__ __launch_bounds__(128) void gemm_wmma_kernel(
    const __bf16* __restrict__ A, const __bf16* __restrict__ Bt,
    void* __restrict__ Cptr) {
  const int tid  = threadIdx.x;
  const int wave = tid >> 5;
  const int lane = tid & 31;
  const int hi   = lane >> 4;
  const int lm   = lane & 15;
  const int mBase = blockIdx.y * 64 + wave * 16;
  const int nBase = blockIdx.x * 64;

#if USE_TDM
  __shared__ __align__(16) __bf16 Btile[2][64][32];
  const bool issuer = (wave == 0);  // wave-uniform; TDM ignores EXEC
  if (issuer)
    tdm_load_b_tile(Bt + (size_t)nBase * D_MODEL,
                    (unsigned int)(uintptr_t)&Btile[0][0][0]);
#endif

  v8f acc[4] = {};
  for (int kb = 0; kb < D_MODEL; kb += 32) {
#if USE_TDM
    const int cur = (kb >> 5) & 1;
    if (issuer) {
      if (kb + 32 < D_MODEL) {
        tdm_load_b_tile(Bt + (size_t)nBase * D_MODEL + (kb + 32),
                        (unsigned int)(uintptr_t)&Btile[cur ^ 1][0][0]);
        __builtin_amdgcn_s_wait_tensorcnt(1);  // current stage complete
      } else {
        __builtin_amdgcn_s_wait_tensorcnt(0);
      }
    }
    __syncthreads();
#endif
    __builtin_prefetch(A + (size_t)(mBase + lm) * D_MODEL + kb + 64, 0, 1);
    v16bf a = frag_a_bf16(A, mBase + lm, D_MODEL, kb, lane);
#pragma unroll
    for (int t = 0; t < 4; ++t) {
#if USE_TDM
      v16bf b = frag_b_bf16(&Btile[cur][0][0], t * 16 + lm, 32, 0, lane);
#else
      v16bf b = frag_b_bf16(Bt, nBase + t * 16 + lm, D_MODEL, kb, lane);
#endif
      acc[t] = wmma_bf16(a, b, acc[t]);
    }
#if USE_TDM
    __syncthreads();  // all waves done reading Btile[cur] before overwrite
#endif
  }

  // C/D layout: lane holds col N = lane&15; VGPR r holds row M = r + 8*hi.
  if (MODE == 2) {
    float* C = (float*)Cptr;
#pragma unroll
    for (int t = 0; t < 4; ++t) {
      const int n = nBase + t * 16 + lm;
#pragma unroll
      for (int r = 0; r < 8; ++r) {
        const int m = mBase + r + 8 * hi;
        C[(size_t)m * D_MODEL + n] = acc[t][r];
      }
    }
  } else {
    __bf16* C = (__bf16*)Cptr;
#pragma unroll
    for (int t = 0; t < 4; ++t) {
      const int n = nBase + t * 16 + lm;
      const int h = n >> 6, d = n & 63;
#pragma unroll
      for (int r = 0; r < 8; ++r) {
        const int m = mBase + r + 8 * hi;
        const int b = m >> 6, l = m & 63;
        const size_t idx =
            (MODE == 0) ? ((((size_t)b * NHEADS + h) * LSEQ + l) * DHEAD + d)
                        : ((((size_t)b * NHEADS + h) * DHEAD + d) * LSEQ + l);
        C[idx] = (__bf16)acc[t][r];
      }
    }
  }
}

// ---- Attention core: one block per (b, h) ---------------------------------
__global__ __launch_bounds__(128) void attn_wmma_kernel(
    const __bf16* __restrict__ q, const __bf16* __restrict__ k,
    const __bf16* __restrict__ vT, const float* __restrict__ bias,
    __bf16* __restrict__ attn) {
  __shared__ __align__(16) float  Ssm[LSEQ][LSEQ + 2];
  __shared__ __align__(16) __bf16 Psm[LSEQ][LSEQ + 8];  // stride 72 (16B mult)

  const int bh = blockIdx.x;
  const int b = bh >> 4, h = bh & 15;
  const int tid  = threadIdx.x;
  const int wave = tid >> 5;
  const int lane = tid & 31;
  const int hi   = lane >> 4;
  const int lm   = lane & 15;
  const int mBase = wave * 16;

  const __bf16* qb = q  + (size_t)bh * LSEQ * DHEAD;
  const __bf16* kb = k  + (size_t)bh * LSEQ * DHEAD;
  const __bf16* vb = vT + (size_t)bh * DHEAD * LSEQ;

  // S = Q @ K^T  (K rows contiguous along Dh -> valid B fragments)
  v8f s[4] = {};
  for (int kk = 0; kk < DHEAD; kk += 32) {
    v16bf a = frag_a_bf16(qb, mBase + lm, DHEAD, kk, lane);
#pragma unroll
    for (int t = 0; t < 4; ++t) {
      v16bf bf = frag_b_bf16(kb, t * 16 + lm, DHEAD, kk, lane);
      s[t] = wmma_bf16(a, bf, s[t]);
    }
  }

  // scale (1/sqrt(64)) + bias, spill to LDS for the row softmax
  const float* biasb = bias + (size_t)h * LSEQ * LSEQ;
#pragma unroll
  for (int t = 0; t < 4; ++t) {
    const int col = t * 16 + lm;
#pragma unroll
    for (int r = 0; r < 8; ++r) {
      const int row = mBase + r + 8 * hi;
      Ssm[row][col] = s[t][r] * 0.125f + biasb[row * LSEQ + col];
    }
  }
  __syncthreads();

  // Row softmax in f32, write P as bf16
  if (tid < LSEQ) {
    float m = -3.402823466e38f;
    for (int c = 0; c < LSEQ; ++c) m = fmaxf(m, Ssm[tid][c]);
    float sum = 0.0f;
    for (int c = 0; c < LSEQ; ++c) sum += __expf(Ssm[tid][c] - m);
    const float inv = 1.0f / sum;
    for (int c = 0; c < LSEQ; ++c)
      Psm[tid][c] = (__bf16)(__expf(Ssm[tid][c] - m) * inv);
  }
  __syncthreads();

  // O = P @ V   (A fragments from LDS via ds_load; V pre-transposed)
  v8f o[4] = {};
  for (int kk = 0; kk < LSEQ; kk += 32) {
    v16bf a = frag_a_bf16(&Psm[0][0], mBase + lm, LSEQ + 8, kk, lane);
#pragma unroll
    for (int t = 0; t < 4; ++t) {
      v16bf bf = frag_b_bf16(vb, t * 16 + lm, LSEQ, kk, lane);
      o[t] = wmma_bf16(a, bf, o[t]);
    }
  }

  // attn buffer layout: [B, L, H*Dh] bf16 (A matrix of the Wo GEMM)
#pragma unroll
  for (int t = 0; t < 4; ++t) {
    const int d = t * 16 + lm;
#pragma unroll
    for (int r = 0; r < 8; ++r) {
      const int row = mBase + r + 8 * hi;  // query position l
      attn[((size_t)b * LSEQ + row) * D_MODEL + h * DHEAD + d] = (__bf16)o[t][r];
    }
  }
}

// ---------------------------------------------------------------------------
extern "C" void kernel_launch(void* const* d_in, const int* in_sizes, int n_in,
                              void* d_out, int out_size, void* d_ws,
                              size_t ws_size, hipStream_t stream) {
  (void)in_sizes; (void)n_in; (void)out_size; (void)ws_size;

  const float* x    = (const float*)d_in[0];
  const float* bias = (const float*)d_in[1];
  const float* Wq   = (const float*)d_in[2];
  const float* Wk   = (const float*)d_in[3];
  const float* Wv   = (const float*)d_in[4];
  const float* Wo   = (const float*)d_in[5];
  float* out = (float*)d_out;

  // Workspace: 4x2MB weights + 64MB xbf + 3x64MB QKV + 64MB attn = 328MB bf16
  const size_t WT_ELEMS  = (size_t)D_MODEL * D_MODEL;
  const size_t X_ELEMS   = (size_t)MROWS * D_MODEL;
  const size_t QKV_ELEMS = (size_t)BATCH * NHEADS * LSEQ * DHEAD;
  char* p = (char*)d_ws;
  __bf16* WqT  = (__bf16*)p; p += WT_ELEMS * sizeof(__bf16);
  __bf16* WkT  = (__bf16*)p; p += WT_ELEMS * sizeof(__bf16);
  __bf16* WvT  = (__bf16*)p; p += WT_ELEMS * sizeof(__bf16);
  __bf16* WoT  = (__bf16*)p; p += WT_ELEMS * sizeof(__bf16);
  __bf16* xbf  = (__bf16*)p; p += X_ELEMS * sizeof(__bf16);
  __bf16* qbuf = (__bf16*)p; p += QKV_ELEMS * sizeof(__bf16);
  __bf16* kbuf = (__bf16*)p; p += QKV_ELEMS * sizeof(__bf16);
  __bf16* vTbuf = (__bf16*)p; p += QKV_ELEMS * sizeof(__bf16);
  __bf16* attnbuf = (__bf16*)p; p += QKV_ELEMS * sizeof(__bf16);

  // 1) weights -> bf16 transposed [N,K]; x -> bf16 (one-time casts)
  dim3 tb(32, 8), tg(D_MODEL / 32, D_MODEL / 32);
  transpose_convert_kernel<<<tg, tb, 0, stream>>>(Wq, WqT);
  transpose_convert_kernel<<<tg, tb, 0, stream>>>(Wk, WkT);
  transpose_convert_kernel<<<tg, tb, 0, stream>>>(Wv, WvT);
  transpose_convert_kernel<<<tg, tb, 0, stream>>>(Wo, WoT);
  convert_x_kernel<<<(unsigned)(X_ELEMS / (256 * 4)), 256, 0, stream>>>(x, xbf);

  // 2) projections: xbf[32768,1024] @ W -> Q/K ([B,H,L,Dh]) and V^T
  dim3 gg(D_MODEL / 64, MROWS / 64);  // (16, 512)
  gemm_wmma_kernel<0><<<gg, 128, 0, stream>>>(xbf, WqT, qbuf);
  gemm_wmma_kernel<0><<<gg, 128, 0, stream>>>(xbf, WkT, kbuf);
  gemm_wmma_kernel<1><<<gg, 128, 0, stream>>>(xbf, WvT, vTbuf);

  // 3) attention: one block per (b, h)
  attn_wmma_kernel<<<BATCH * NHEADS, 128, 0, stream>>>(qbuf, kbuf, vTbuf,
                                                       bias, attnbuf);

  // 4) output projection: attn[32768,1024] @ Wo -> f32 out
  gemm_wmma_kernel<2><<<gg, 128, 0, stream>>>(attnbuf, WoT, out);
}
